// MulPQNet_53738630808194
// MI455X (gfx1250) — matrix operand order repaired and checked by reading, not compiled, not run on hardware
//
#include <hip/hip_runtime.h>
#include <hip/hip_bf16.h>
#include <math.h>

// ---------------------------------------------------------------------------
// CDNA5 (gfx1250) wave32 WMMA types
// ---------------------------------------------------------------------------
typedef _Float16 v16h __attribute__((ext_vector_type(16)));
typedef _Float16 h8   __attribute__((ext_vector_type(8)));
typedef float    v8f  __attribute__((ext_vector_type(8)));

static __device__ __forceinline__ v16h cat8(h8 a, h8 b) {
  return __builtin_shufflevector(a, b, 0,1,2,3,4,5,6,7,8,9,10,11,12,13,14,15);
}
static __device__ __forceinline__ h8 cvt8(float4 u0, float4 u1) {
  h8 r;
  r[0]=(_Float16)u0.x; r[1]=(_Float16)u0.y; r[2]=(_Float16)u0.z; r[3]=(_Float16)u0.w;
  r[4]=(_Float16)u1.x; r[5]=(_Float16)u1.y; r[6]=(_Float16)u1.z; r[7]=(_Float16)u1.w;
  return r;
}

// ---------------------------------------------------------------------------
// WMMA GEMM:  C[M,N] = act(A[M,K] * B[N,K]^T + bias[N])
// f32 operands, cvt->f16 in-register, double-buffered LDS, 64x128 tile,
// 8 waves (4x2), each wave 16x64 via 4x v_wmma_f32_16x16x32_f16 per K-step.
// Requires K % 32 == 0 (all call sites: 9216/4096/4096/128).
// Row indices are CLAMPED on load (garbage only reaches unstored rows/cols).
// Pipeline: raw global loads issued BEFORE the WMMA block; cvt + ds_store
// AFTER it, so s_wait_loadcnt sits behind the matrix math.
// ACT: 0 = none, 1 = relu, 2 = leaky(0.01)
// ---------------------------------------------------------------------------
template <int ACT>
__global__ __launch_bounds__(256) void gemm_wmma(
    const float* __restrict__ A, const float* __restrict__ Bm,
    const float* __restrict__ bias, float* __restrict__ C,
    int M, int N, int K)
{
  __shared__ h8   AsQ[2][64][4];    // 8 KB
  __shared__ v16h BsV[2][128][2];   // 16 KB

  const int t    = threadIdx.x;
  const int lane = t & 31, wave = t >> 5;
  const int wm   = wave >> 1, wn = wave & 1;
  const int lm   = lane & 15, lh = lane >> 4;
  const int m0   = blockIdx.y * 64, n0 = blockIdx.x * 128;

  const int rowA = t >> 2, kkA = (t & 3) * 8;    // 64 rows x 32 halves
  const int rowB = t >> 1, kkB = (t & 1) * 16;   // 128 rows x 32 halves
  int gmA = m0 + rowA; if (gmA >= M) gmA = M - 1;
  int gnB = n0 + rowB; if (gnB >= N) gnB = N - 1;
  const float* apA = A  + (size_t)gmA * K + kkA;
  const float* apB = Bm + (size_t)gnB * K + kkB;

  _Float16* myA = (_Float16*)AsQ[0] + rowA * 32 + kkA;   // buffer stride below
  _Float16* myB = (_Float16*)BsV[0] + rowB * 32 + kkB;
  const int strideA = 64 * 32;    // halves per A buffer
  const int strideB = 128 * 32;   // halves per B buffer

  v8f acc[4] = {};
  const int nsteps = K >> 5;

  // prologue: stage K-tile 0
  {
    const float4* p = (const float4*)apA;
    const float4* q = (const float4*)apB;
    float4 a0 = p[0], a1 = p[1];
    float4 b0 = q[0], b1 = q[1], b2 = q[2], b3 = q[3];
    *(h8*)myA = cvt8(a0, a1);
    *(h8*)myB = cvt8(b0, b1);
    *(h8*)(myB + 8) = cvt8(b2, b3);
  }
  __syncthreads();

  for (int s = 0; s < nsteps - 1; ++s) {
    const int cur = s & 1, nxt = cur ^ 1;

    // issue raw loads of the NEXT K-tile (no waits yet)
    const float4* p = (const float4*)(apA + (s + 1) * 32);
    const float4* q = (const float4*)(apB + (s + 1) * 32);
    float4 a0 = p[0], a1 = p[1];
    float4 b0 = q[0], b1 = q[1], b2 = q[2], b3 = q[3];

    // fragments (load all five, then 4 WMMAs)
    v16h av  = cat8(AsQ[cur][wm * 16 + lm][2 * lh], AsQ[cur][wm * 16 + lm][2 * lh + 1]);
    v16h bv0 = BsV[cur][wn * 64 +  0 + lm][lh];
    v16h bv1 = BsV[cur][wn * 64 + 16 + lm][lh];
    v16h bv2 = BsV[cur][wn * 64 + 32 + lm][lh];
    v16h bv3 = BsV[cur][wn * 64 + 48 + lm][lh];
    acc[0] = __builtin_amdgcn_wmma_f32_16x16x32_f16(false, av, false, bv0, (short)0, acc[0], false, false);
    acc[1] = __builtin_amdgcn_wmma_f32_16x16x32_f16(false, av, false, bv1, (short)0, acc[1], false, false);
    acc[2] = __builtin_amdgcn_wmma_f32_16x16x32_f16(false, av, false, bv2, (short)0, acc[2], false, false);
    acc[3] = __builtin_amdgcn_wmma_f32_16x16x32_f16(false, av, false, bv3, (short)0, acc[3], false, false);

    // convert + commit prefetched tile (load-wait lands here, behind WMMAs)
    *(h8*)(myA + nxt * strideA) = cvt8(a0, a1);
    *(h8*)(myB + nxt * strideB) = cvt8(b0, b1);
    *(h8*)(myB + nxt * strideB + 8) = cvt8(b2, b3);
    __syncthreads();
  }

  // final K-step (no prefetch)
  {
    const int cur = (nsteps - 1) & 1;
    v16h av  = cat8(AsQ[cur][wm * 16 + lm][2 * lh], AsQ[cur][wm * 16 + lm][2 * lh + 1]);
    v16h bv0 = BsV[cur][wn * 64 +  0 + lm][lh];
    v16h bv1 = BsV[cur][wn * 64 + 16 + lm][lh];
    v16h bv2 = BsV[cur][wn * 64 + 32 + lm][lh];
    v16h bv3 = BsV[cur][wn * 64 + 48 + lm][lh];
    acc[0] = __builtin_amdgcn_wmma_f32_16x16x32_f16(false, av, false, bv0, (short)0, acc[0], false, false);
    acc[1] = __builtin_amdgcn_wmma_f32_16x16x32_f16(false, av, false, bv1, (short)0, acc[1], false, false);
    acc[2] = __builtin_amdgcn_wmma_f32_16x16x32_f16(false, av, false, bv2, (short)0, acc[2], false, false);
    acc[3] = __builtin_amdgcn_wmma_f32_16x16x32_f16(false, av, false, bv3, (short)0, acc[3], false, false);
  }

#pragma unroll
  for (int f = 0; f < 4; ++f) {
    const int n = n0 + wn * 64 + f * 16 + lm;
    const float bb = (bias != nullptr && n < N) ? bias[n] : 0.0f;
#pragma unroll
    for (int i = 0; i < 8; ++i) {
      const int m = m0 + wm * 16 + (lh << 3) + i;
      float v = acc[f][i] + bb;
      if (ACT == 1) v = fmaxf(v, 0.0f);
      if (ACT == 2) v = v > 0.0f ? v : 0.01f * v;
      if (m < M && n < N) C[(size_t)m * N + n] = v;
    }
  }
}

// ---------------------------------------------------------------------------
// Implicit-GEMM convolution (NCHW), fused bias + ReLU.
// M = Cout, N = NB*OH*OW, K = Cin*KH*KW (compile-time geometry).
// Two-phase gather: phase 1 issues 16+8 raw clamped-address loads,
// phase 2 (after the WMMA block) converts/selects/stores to LDS.
// ---------------------------------------------------------------------------
template <int Cin, int IH, int IW, int Cout, int KH, int KW, int S, int P, int OH, int OW>
__global__ __launch_bounds__(256) void conv_igemm(
    const float* __restrict__ in, const float* __restrict__ w,
    const float* __restrict__ bias, float* __restrict__ out, int NB)
{
  constexpr int Ktot = Cin * KH * KW;
  constexpr int KHW  = KH * KW;
  constexpr int OHW  = OH * OW;

  __shared__ h8   AsQ[2][64][4];
  __shared__ v16h BsV[2][128][2];

  const int t    = threadIdx.x;
  const int lane = t & 31, wave = t >> 5;
  const int wm   = wave >> 1, wn = wave & 1;
  const int lm   = lane & 15, lh = lane >> 4;
  const int m0   = blockIdx.y * 64, n0 = blockIdx.x * 128;
  const int Ntot = NB * OHW;

  const int rowA = t >> 2, kkA = (t & 3) * 8;
  const int rowB = t >> 1, kkB = (t & 1) * 16;

  int gmA = m0 + rowA; if (gmA >= Cout) gmA = Cout - 1;
  const float* wrow = w + (size_t)gmA * Ktot;

  int gnB = n0 + rowB; if (gnB >= Ntot) gnB = Ntot - 1;
  const int bb_ = gnB / OHW;
  const int pp_ = gnB % OHW;
  const int oh_ = pp_ / OW;
  const int ow_ = pp_ % OW;
  const float* ib = in + (size_t)bb_ * Cin * IH * IW;

  _Float16* myA = (_Float16*)AsQ[0] + rowA * 32 + kkA;
  _Float16* myB = (_Float16*)BsV[0] + rowB * 32 + kkB;
  const int strideA = 64 * 32;
  const int strideB = 128 * 32;

  // phase-1: issue raw loads (clamped, unconditional)
  float arw[8]; bool aok[8];
  float brw[16]; bool bok[16];
  auto issueA = [&](int k0) {
#pragma unroll
    for (int j = 0; j < 8; ++j) {
      const int k  = k0 + kkA + j;
      const int kc = (k < Ktot) ? k : (Ktot - 1);
      aok[j] = (k < Ktot);
      arw[j] = wrow[kc];
    }
  };
  auto issueB = [&](int k0) {
#pragma unroll
    for (int j = 0; j < 16; ++j) {
      const int k  = k0 + kkB + j;
      const int kc = (k < Ktot) ? k : (Ktot - 1);
      const int ci = kc / KHW;
      const int rs = kc % KHW;
      const int r  = rs / KW;
      const int ss = rs % KW;
      const int ih = oh_ * S - P + r;
      const int iw = ow_ * S - P + ss;
      bok[j] = (k < Ktot) & ((unsigned)ih < (unsigned)IH) & ((unsigned)iw < (unsigned)IW);
      const int ihc = ih < 0 ? 0 : (ih >= IH ? IH - 1 : ih);
      const int iwc = iw < 0 ? 0 : (iw >= IW ? IW - 1 : iw);
      brw[j] = ib[((size_t)ci * IH + ihc) * IW + iwc];
    }
  };
  // phase-2: convert + select + commit to LDS buffer `buf`
  auto commit = [&](int buf) {
    h8 ra, r0, r1;
#pragma unroll
    for (int j = 0; j < 8; ++j) ra[j] = (_Float16)(aok[j] ? arw[j] : 0.0f);
#pragma unroll
    for (int j = 0; j < 8; ++j) r0[j] = (_Float16)(bok[j] ? brw[j] : 0.0f);
#pragma unroll
    for (int j = 0; j < 8; ++j) r1[j] = (_Float16)(bok[j + 8] ? brw[j + 8] : 0.0f);
    *(h8*)(myA + buf * strideA) = ra;
    *(h8*)(myB + buf * strideB) = r0;
    *(h8*)(myB + buf * strideB + 8) = r1;
  };

  v8f acc[4] = {};
  const int nsteps = (Ktot + 31) >> 5;

  issueA(0); issueB(0); commit(0);
  __syncthreads();

  for (int s = 0; s < nsteps - 1; ++s) {
    const int cur = s & 1, nxt = cur ^ 1;
    issueA((s + 1) * 32);
    issueB((s + 1) * 32);

    v16h av  = cat8(AsQ[cur][wm * 16 + lm][2 * lh], AsQ[cur][wm * 16 + lm][2 * lh + 1]);
    v16h bv0 = BsV[cur][wn * 64 +  0 + lm][lh];
    v16h bv1 = BsV[cur][wn * 64 + 16 + lm][lh];
    v16h bv2 = BsV[cur][wn * 64 + 32 + lm][lh];
    v16h bv3 = BsV[cur][wn * 64 + 48 + lm][lh];
    acc[0] = __builtin_amdgcn_wmma_f32_16x16x32_f16(false, av, false, bv0, (short)0, acc[0], false, false);
    acc[1] = __builtin_amdgcn_wmma_f32_16x16x32_f16(false, av, false, bv1, (short)0, acc[1], false, false);
    acc[2] = __builtin_amdgcn_wmma_f32_16x16x32_f16(false, av, false, bv2, (short)0, acc[2], false, false);
    acc[3] = __builtin_amdgcn_wmma_f32_16x16x32_f16(false, av, false, bv3, (short)0, acc[3], false, false);

    commit(nxt);
    __syncthreads();
  }

  {
    const int cur = (nsteps - 1) & 1;
    v16h av  = cat8(AsQ[cur][wm * 16 + lm][2 * lh], AsQ[cur][wm * 16 + lm][2 * lh + 1]);
    v16h bv0 = BsV[cur][wn * 64 +  0 + lm][lh];
    v16h bv1 = BsV[cur][wn * 64 + 16 + lm][lh];
    v16h bv2 = BsV[cur][wn * 64 + 32 + lm][lh];
    v16h bv3 = BsV[cur][wn * 64 + 48 + lm][lh];
    acc[0] = __builtin_amdgcn_wmma_f32_16x16x32_f16(false, av, false, bv0, (short)0, acc[0], false, false);
    acc[1] = __builtin_amdgcn_wmma_f32_16x16x32_f16(false, av, false, bv1, (short)0, acc[1], false, false);
    acc[2] = __builtin_amdgcn_wmma_f32_16x16x32_f16(false, av, false, bv2, (short)0, acc[2], false, false);
    acc[3] = __builtin_amdgcn_wmma_f32_16x16x32_f16(false, av, false, bv3, (short)0, acc[3], false, false);
  }

  // epilogue: bias + ReLU, scatter to NCHW
#pragma unroll
  for (int f = 0; f < 4; ++f) {
    const int n = n0 + wn * 64 + f * 16 + lm;
#pragma unroll
    for (int i = 0; i < 8; ++i) {
      const int m = m0 + wm * 16 + (lh << 3) + i;
      if (m < Cout && n < Ntot) {
        const float v = fmaxf(acc[f][i] + bias[m], 0.0f);
        const int b = n / OHW, pp = n % OHW;
        out[((size_t)b * Cout + m) * OHW + pp] = v;
      }
    }
  }
}

// ---------------------------------------------------------------------------
// 3x3 stride-2 VALID maxpool, NCHW
// ---------------------------------------------------------------------------
__global__ __launch_bounds__(256) void maxpool3s2(
    const float* __restrict__ in, float* __restrict__ out,
    int C, int IH, int IW, int OH, int OW, int total)
{
  int idx = blockIdx.x * blockDim.x + threadIdx.x;
  if (idx >= total) return;
  int ow = idx % OW; int u = idx / OW;
  int oh = u % OH;   u /= OH;
  int c  = u % C;    int b = u / C;
  const float* p = in + ((size_t)b * C + c) * IH * IW;
  float m = -INFINITY;
#pragma unroll
  for (int r = 0; r < 3; ++r)
#pragma unroll
    for (int s = 0; s < 3; ++s)
      m = fmaxf(m, p[(oh * 2 + r) * IW + (ow * 2 + s)]);
  out[idx] = m;
}

// ---------------------------------------------------------------------------
// PQ head: per (b,l) block. Codebook slab (256x32 f32 = 32KB) staged in LDS.
// ---------------------------------------------------------------------------
__global__ __launch_bounds__(256) void pq_head(
    const float* __restrict__ h, const float* __restrict__ cb,
    float* __restrict__ hardf, float* __restrict__ softf)
{
  constexpr int PK = 256, PS = 32;
  const int l = blockIdx.x, b = blockIdx.y, t = threadIdx.x;

  __shared__ float scb[PK * PS];
  __shared__ float sx[PS];
  __shared__ float pr[PK];
  __shared__ float red[PK];
  __shared__ int   redi[PK];
  __shared__ float ssoft[PS];

  const float* cbl = cb + (size_t)l * PK * PS;
#pragma unroll 4
  for (int j = 0; j < PS; ++j) scb[t * PS + j] = cbl[t * PS + j];
  if (t < PS) sx[t] = h[b * 128 + l * PS + t];
  __syncthreads();

  float xn2 = 0.0f;
#pragma unroll
  for (int d = 0; d < PS; ++d) xn2 += sx[d] * sx[d];
  float dot = 0.0f, cn2 = 0.0f;
#pragma unroll 8
  for (int d = 0; d < PS; ++d) {
    float c = scb[t * PS + d];
    dot += sx[d] * c; cn2 += c * c;
  }
  const float sim = dot / fmaxf(sqrtf(xn2) * sqrtf(cn2), 1e-8f);

  red[t] = sim; redi[t] = t;
  __syncthreads();
  for (int s = 128; s > 0; s >>= 1) {
    if (t < s) {
      float o = red[t + s]; int oi = redi[t + s];
      if (o > red[t] || (o == red[t] && oi < redi[t])) { red[t] = o; redi[t] = oi; }
    }
    __syncthreads();
  }
  const float smax = red[0];
  const int   imax = redi[0];
  __syncthreads();

  const float e = __expf(20.0f * (sim - smax));
  pr[t] = e; red[t] = e;
  __syncthreads();
  for (int s = 128; s > 0; s >>= 1) {
    if (t < s) red[t] += red[t + s];
    __syncthreads();
  }
  const float inv = 1.0f / red[0];
  __syncthreads();

  if (t < PS) {
    float s = 0.0f;
    for (int k = 0; k < PK; ++k) s += pr[k] * scb[k * PS + t];
    ssoft[t] = s * inv;
  }
  __syncthreads();

  if (t < PS) {
    float n2 = 0.0f;
#pragma unroll
    for (int d = 0; d < PS; ++d) n2 += ssoft[d] * ssoft[d];
    softf[b * 128 + l * PS + t] = ssoft[t] / fmaxf(sqrtf(n2), 1e-12f);
    float h2 = 0.0f;
#pragma unroll
    for (int d = 0; d < PS; ++d) { float c = scb[imax * PS + d]; h2 += c * c; }
    hardf[b * 128 + l * PS + t] = scb[imax * PS + t] / fmaxf(sqrtf(h2), 1e-12f);
  }
}

// ---------------------------------------------------------------------------
// Launch pipeline
// ---------------------------------------------------------------------------
static inline int cdiv_i(int a, int b) { return (a + b - 1) / b; }

extern "C" void kernel_launch(void* const* d_in, const int* in_sizes, int n_in,
                              void* d_out, int out_size, void* d_ws, size_t ws_size,
                              hipStream_t stream) {
  (void)n_in; (void)out_size; (void)ws_size;
  const float* x   = (const float*)d_in[0];
  const float* w1  = (const float*)d_in[1];
  const float* b1  = (const float*)d_in[2];
  const float* w2  = (const float*)d_in[3];
  const float* b2  = (const float*)d_in[4];
  const float* w3  = (const float*)d_in[5];
  const float* b3  = (const float*)d_in[6];
  const float* w4  = (const float*)d_in[7];
  const float* b4  = (const float*)d_in[8];
  const float* w5  = (const float*)d_in[9];
  const float* b5  = (const float*)d_in[10];
  const float* cw1 = (const float*)d_in[11];
  const float* cb1 = (const float*)d_in[12];
  const float* cw2 = (const float*)d_in[13];
  const float* cb2 = (const float*)d_in[14];
  const float* fw  = (const float*)d_in[15];
  const float* fb  = (const float*)d_in[16];
  const float* cbk = (const float*)d_in[17];
  const float* rw  = (const float*)d_in[18];
  float* out = (float*)d_out;

  const int NB = in_sizes[0] / (3 * 224 * 224);   // batch (128)

  float* ws   = (float*)d_ws;
  const size_t sz0 = (size_t)NB * 64 * 55 * 55;   // conv1 out (largest)
  const size_t sz1 = (size_t)NB * 384 * 13 * 13;  // conv3 out
  float* buf0 = ws;
  float* buf1 = ws + sz0;
  float* buf2 = buf1 + sz1;

  conv_igemm<3, 224, 224, 64, 11, 11, 4, 2, 55, 55>
      <<<dim3(cdiv_i(NB * 55 * 55, 128), 1), 256, 0, stream>>>(x, w1, b1, buf0, NB);
  { int tot = NB * 64 * 27 * 27;
    maxpool3s2<<<cdiv_i(tot, 256), 256, 0, stream>>>(buf0, buf1, 64, 55, 55, 27, 27, tot); }
  conv_igemm<64, 27, 27, 192, 5, 5, 1, 2, 27, 27>
      <<<dim3(cdiv_i(NB * 27 * 27, 128), 3), 256, 0, stream>>>(buf1, w2, b2, buf2, NB);
  { int tot = NB * 192 * 13 * 13;
    maxpool3s2<<<cdiv_i(tot, 256), 256, 0, stream>>>(buf2, buf0, 192, 27, 27, 13, 13, tot); }
  conv_igemm<192, 13, 13, 384, 3, 3, 1, 1, 13, 13>
      <<<dim3(cdiv_i(NB * 13 * 13, 128), 6), 256, 0, stream>>>(buf0, w3, b3, buf1, NB);
  conv_igemm<384, 13, 13, 256, 3, 3, 1, 1, 13, 13>
      <<<dim3(cdiv_i(NB * 13 * 13, 128), 4), 256, 0, stream>>>(buf1, w4, b4, buf2, NB);
  conv_igemm<256, 13, 13, 256, 3, 3, 1, 1, 13, 13>
      <<<dim3(cdiv_i(NB * 13 * 13, 128), 4), 256, 0, stream>>>(buf2, w5, b5, buf0, NB);
  { int tot = NB * 256 * 6 * 6;
    maxpool3s2<<<cdiv_i(tot, 256), 256, 0, stream>>>(buf0, buf1, 256, 13, 13, 6, 6, tot); }

  gemm_wmma<1><<<dim3(cdiv_i(4096, 128), cdiv_i(NB, 64)), 256, 0, stream>>>(
      buf1, cw1, cb1, buf2, NB, 4096, 9216);
  gemm_wmma<1><<<dim3(cdiv_i(4096, 128), cdiv_i(NB, 64)), 256, 0, stream>>>(
      buf2, cw2, cb2, buf0, NB, 4096, 4096);
  gemm_wmma<0><<<dim3(cdiv_i(128, 128), cdiv_i(NB, 64)), 256, 0, stream>>>(
      buf0, fw, fb, out, NB, 128, 4096);

  pq_head<<<dim3(4, NB), 256, 0, stream>>>(
      out, cbk, out + (size_t)NB * 128, out + (size_t)NB * 256);

  gemm_wmma<2><<<dim3(cdiv_i(300, 128), cdiv_i(NB, 64)), 256, 0, stream>>>(
      out + (size_t)NB * 256, rw, nullptr, out + (size_t)NB * 384, NB, 300, 128);
}
